// HAN_29059748725078
// MI455X (gfx1250) — matrix operand order repaired and checked by reading, not compiled, not run on hardware
//
#include <hip/hip_runtime.h>
#include <hip/hip_bf16.h>

typedef __attribute__((ext_vector_type(16))) __bf16 v16bf;
typedef __attribute__((ext_vector_type(8)))  __bf16 v8bf;
typedef __attribute__((ext_vector_type(8)))  float  v8f;
typedef __attribute__((ext_vector_type(4)))  float  v4f;
typedef __attribute__((ext_vector_type(4)))  int    v4i;

#define HAN_N 4096
#define HAN_F 512
#define HAN_D 64
#define HAN_H 4
#define HAN_M 2
#define HAN_ALPHA 0.2f

// ---------------- workspace layout (bytes) ----------------
#define OFF_XB    0u                         // [N][F]   bf16  : 4 MB
#define OFF_WTB   4194304u                   // [H][D][F] bf16 : 256 KB
#define OFF_WHT   4456448u                   // [H][D][N] bf16 : 2 MB
#define OFF_WHF   6553600u                   // [H][N][D] f32  : 4 MB
#define OFF_S1    10747904u                  // [H][N]   f32   : 64 KB
#define OFF_S2    10813440u                  // [H][N]   f32   : 64 KB
#define OFF_S2MAX 10878976u                  // [H]      f32

__device__ __forceinline__ float leaky(float t) {
    return t > 0.f ? t : HAN_ALPHA * t;
}

// -------- x -> bf16 --------
__global__ void k_cvt_x(const float* __restrict__ x, __bf16* __restrict__ xb) {
    int i = blockIdx.x * blockDim.x + threadIdx.x;
    if (i < HAN_N * HAN_F) xb[i] = (__bf16)x[i];
}

// -------- W [H][F][D] f32 -> WTb [H][D][F] bf16 (transposed) --------
__global__ void k_tr_w(const float* __restrict__ W, __bf16* __restrict__ WTb) {
    int i = blockIdx.x * blockDim.x + threadIdx.x;
    if (i >= HAN_H * HAN_F * HAN_D) return;
    int d = i % HAN_D;
    int f = (i / HAN_D) % HAN_F;
    int h = i / (HAN_F * HAN_D);
    WTb[(h * HAN_D + d) * HAN_F + f] = (__bf16)W[i];
}

// -------- Wh = x @ W per head, via WMMA bf16.  One wave = 16 rows x 64 cols. --------
__global__ void k_wh(const __bf16* __restrict__ xb, const __bf16* __restrict__ WTb,
                     __bf16* __restrict__ WhT, float* __restrict__ Whf) {
    const int lane = threadIdx.x & 31;
    const int wv   = threadIdx.x >> 5;
    const int tile = blockIdx.x * 4 + wv;          // 0 .. H*(N/16)-1
    const int h    = tile >> 8;                    // tile / 256
    const int ib   = (tile & 255) << 4;            // row base
    const int hf   = lane >> 4;                    // half-wave
    const int lm   = lane & 15;

    v8f acc[4];
    #pragma unroll
    for (int t = 0; t < 4; ++t)
        #pragma unroll
        for (int v = 0; v < 8; ++v) acc[t][v] = 0.f;

    const __bf16* arow = xb + (size_t)(ib + lm) * HAN_F;
    for (int kb = 0; kb < HAN_F; kb += 32) {
        // A fragment: 16x32 bf16.  elems 0..7 -> K = hf*16+e ; 8..15 -> K = hf*16+8+e
        v8bf g0 = *(const v8bf*)(arow + kb + hf * 16);
        v8bf g1 = *(const v8bf*)(arow + kb + hf * 16 + 8);
        v16bf a;
        #pragma unroll
        for (int e = 0; e < 8; ++e) { a[e] = g0[e]; a[e + 8] = g1[e]; }
        #pragma unroll
        for (int t = 0; t < 4; ++t) {
            // B fragment: 32x16.  lane<16: K 0..15 ; lane>=16: K 16..31 ; N = lm
            const __bf16* bp = WTb + (size_t)(h * HAN_D + t * 16 + lm) * HAN_F + kb + hf * 16;
            v16bf b = *(const v16bf*)bp;
            acc[t] = __builtin_amdgcn_wmma_f32_16x16x32_bf16(
                false, a, false, b, (short)0, acc[t], false, false);
        }
    }
    // C layout: elem v -> row M = v + hf*8, col N = lm
    #pragma unroll
    for (int t = 0; t < 4; ++t)
        #pragma unroll
        for (int v = 0; v < 8; ++v) {
            int n = ib + hf * 8 + v;
            int d = t * 16 + lm;
            float val = acc[t][v];
            Whf[((size_t)h * HAN_N + n) * HAN_D + d] = val;
            WhT[((size_t)h * HAN_D + d) * HAN_N + n] = (__bf16)val;
        }
}

// -------- s1/s2 = Wh . a1/a2 --------
__global__ void k_s(const float* __restrict__ Whf, const float* __restrict__ a1,
                    const float* __restrict__ a2, float* __restrict__ s1,
                    float* __restrict__ s2) {
    int i = blockIdx.x * blockDim.x + threadIdx.x;   // i = h*N + n
    if (i >= HAN_H * HAN_N) return;
    int h = i / HAN_N;
    const float* wr = Whf + (size_t)i * HAN_D;
    const float* p1 = a1 + h * HAN_D;
    const float* p2 = a2 + h * HAN_D;
    float r1 = 0.f, r2 = 0.f;
    for (int d = 0; d < HAN_D; ++d) { float w = wr[d]; r1 += w * p1[d]; r2 += w * p2[d]; }
    s1[i] = r1; s2[i] = r2;
}

// -------- per-head max of s2 (softmax shift bound) --------
__global__ void k_s2max(const float* __restrict__ s2, float* __restrict__ s2max) {
    __shared__ float red[256];
    int h = blockIdx.x;
    float m = -1e30f;
    for (int n = threadIdx.x; n < HAN_N; n += 256) m = fmaxf(m, s2[h * HAN_N + n]);
    red[threadIdx.x] = m;
    __syncthreads();
    for (int s = 128; s > 0; s >>= 1) {
        if (threadIdx.x < (unsigned)s) red[threadIdx.x] = fmaxf(red[threadIdx.x], red[threadIdx.x + s]);
        __syncthreads();
    }
    if (threadIdx.x == 0) s2max[h] = red[0];
}

// -------- fused masked-softmax attention + att@Wh + elu + sum over metapaths --------
// One wave = (head h, 16-row block).  j-outer loop; both metapaths share one set of
// exp() scores (mask is the only per-m difference) and one set of B fragments.
// Row sums computed on the matrix pipe via an all-ones B matrix: D = A x 1 puts each
// row's sum in every column, landing exactly in the lane/elem that needs it.
__global__ void k_attn(const int* __restrict__ adj, const __bf16* __restrict__ WhT,
                       const float* __restrict__ s1, const float* __restrict__ s2,
                       const float* __restrict__ s2max, float* __restrict__ out) {
    const int lane = threadIdx.x & 31;
    const int wv   = threadIdx.x >> 5;
    const int tile = blockIdx.x * 4 + wv;
    const int h    = tile >> 8;
    const int ib   = (tile & 255) << 4;
    const int hf   = lane >> 4;
    const int lm   = lane & 15;
    const int row  = ib + lm;                       // row handled for score generation

    const float  s1v = s1[h * HAN_N + row];
    const float  C   = leaky(s1v + s2max[h]);       // >= every e in this row (leaky monotone)
    const float* s2h = s2 + h * HAN_N;

    v16bf ones;
    #pragma unroll
    for (int e = 0; e < 16; ++e) ones[e] = (__bf16)1.0f;
    const __bf16 zb = (__bf16)0.0f;

    v8f acc0[4], acc1[4];
    v8f accs0, accs1;                               // row-sum accumulators (A x ones)
    #pragma unroll
    for (int t = 0; t < 4; ++t)
        #pragma unroll
        for (int v = 0; v < 8; ++v) { acc0[t][v] = 0.f; acc1[t][v] = 0.f; }
    #pragma unroll
    for (int v = 0; v < 8; ++v) { accs0[v] = 0.f; accs1[v] = 0.f; }

    const int* arow0 = adj + (size_t)row * HAN_N;                         // m = 0
    const int* arow1 = adj + ((size_t)HAN_N + row) * HAN_N;               // m = 1

    for (int jb = 0; jb < HAN_N; jb += 32) {
        const int c0 = jb + hf * 16;                // this lane's column base
        v4f u0 = *(const v4f*)(s2h + c0);
        v4f u1 = *(const v4f*)(s2h + c0 + 4);
        v4f u2 = *(const v4f*)(s2h + c0 + 8);
        v4f u3 = *(const v4f*)(s2h + c0 + 12);
        v4i q0 = *(const v4i*)(arow0 + c0);
        v4i q1 = *(const v4i*)(arow0 + c0 + 4);
        v4i q2 = *(const v4i*)(arow0 + c0 + 8);
        v4i q3 = *(const v4i*)(arow0 + c0 + 12);
        v4i r0 = *(const v4i*)(arow1 + c0);
        v4i r1 = *(const v4i*)(arow1 + c0 + 4);
        v4i r2 = *(const v4i*)(arow1 + c0 + 8);
        v4i r3 = *(const v4i*)(arow1 + c0 + 12);

        // A fragments: elems 0..15 -> cols c0+0..15
        v16bf a0, a1;
        #pragma unroll
        for (int e = 0; e < 4; ++e) {
            __bf16 pb;
            pb = (__bf16)__expf(leaky(s1v + u0[e]) - C);
            a0[e]      = q0[e] > 0 ? pb : zb;  a1[e]      = r0[e] > 0 ? pb : zb;
            pb = (__bf16)__expf(leaky(s1v + u1[e]) - C);
            a0[e + 4]  = q1[e] > 0 ? pb : zb;  a1[e + 4]  = r1[e] > 0 ? pb : zb;
            pb = (__bf16)__expf(leaky(s1v + u2[e]) - C);
            a0[e + 8]  = q2[e] > 0 ? pb : zb;  a1[e + 8]  = r2[e] > 0 ? pb : zb;
            pb = (__bf16)__expf(leaky(s1v + u3[e]) - C);
            a0[e + 12] = q3[e] > 0 ? pb : zb;  a1[e + 12] = r3[e] > 0 ? pb : zb;
        }

        #pragma unroll
        for (int t = 0; t < 4; ++t) {
            const __bf16* bp = WhT + (size_t)(h * HAN_D + t * 16 + lm) * HAN_N + jb + hf * 16;
            v16bf b = *(const v16bf*)bp;
            acc0[t] = __builtin_amdgcn_wmma_f32_16x16x32_bf16(
                false, a0, false, b, (short)0, acc0[t], false, false);
            acc1[t] = __builtin_amdgcn_wmma_f32_16x16x32_bf16(
                false, a1, false, b, (short)0, acc1[t], false, false);
        }
        accs0 = __builtin_amdgcn_wmma_f32_16x16x32_bf16(
            false, a0, false, ones, (short)0, accs0, false, false);
        accs1 = __builtin_amdgcn_wmma_f32_16x16x32_bf16(
            false, a1, false, ones, (short)0, accs1, false, false);
    }

    // C-fragment elem v is row hf*8+v; accs*[v] holds that row's sum in this lane.
    #pragma unroll
    for (int v = 0; v < 8; ++v) {
        float inv0 = 1.0f / fmaxf(accs0[v], 1e-30f);
        float inv1 = 1.0f / fmaxf(accs1[v], 1e-30f);
        int n = ib + hf * 8 + v;
        #pragma unroll
        for (int t = 0; t < 4; ++t) {
            float x0 = acc0[t][v] * inv0;           // softmax-normalized att @ Wh, m=0
            float x1 = acc1[t][v] * inv1;           // m=1
            float r = ((x0 > 0.f) ? x0 : (__expf(x0) - 1.0f))
                    + ((x1 > 0.f) ? x1 : (__expf(x1) - 1.0f));   // elu + sum over m (beta==1)
            int d = t * 16 + lm;
            out[(size_t)n * (HAN_H * HAN_D) + h * HAN_D + d] = r;
        }
    }
}

extern "C" void kernel_launch(void* const* d_in, const int* in_sizes, int n_in,
                              void* d_out, int out_size, void* d_ws, size_t ws_size,
                              hipStream_t stream) {
    (void)in_sizes; (void)n_in; (void)out_size; (void)ws_size;
    const float* x   = (const float*)d_in[0];
    const int*   adj = (const int*)d_in[1];
    const float* W   = (const float*)d_in[2];
    const float* a1  = (const float*)d_in[3];
    const float* a2  = (const float*)d_in[4];
    // d_in[5..7] (Ws, bs, q_sem) provably do not affect the output:
    // softmax over the singleton axis of [M,1] makes beta == 1.

    char* ws = (char*)d_ws;
    __bf16* xb    = (__bf16*)(ws + OFF_XB);
    __bf16* WTb   = (__bf16*)(ws + OFF_WTB);
    __bf16* WhT   = (__bf16*)(ws + OFF_WHT);
    float*  Whf   = (float*)(ws + OFF_WHF);
    float*  s1    = (float*)(ws + OFF_S1);
    float*  s2    = (float*)(ws + OFF_S2);
    float*  s2max = (float*)(ws + OFF_S2MAX);
    float*  out   = (float*)d_out;

    k_cvt_x<<<(HAN_N * HAN_F + 255) / 256, 256, 0, stream>>>(x, xb);
    k_tr_w<<<(HAN_H * HAN_F * HAN_D + 255) / 256, 256, 0, stream>>>(W, WTb);
    k_wh<<<(HAN_H * (HAN_N / 16)) / 4, 128, 0, stream>>>(xb, WTb, WhT, Whf);
    k_s<<<(HAN_H * HAN_N + 255) / 256, 256, 0, stream>>>(Whf, a1, a2, s1, s2);
    k_s2max<<<HAN_H, 256, 0, stream>>>(s2, s2max);
    k_attn<<<(HAN_H * (HAN_N / 16)) / 4, 128, 0, stream>>>(adj, WhT, s1, s2, s2max, out);
}
